// Ours1_WeightedHS_67422396613013
// MI455X (gfx1250) — matrix operand order repaired and checked by reading, not compiled, not run on hardware
//
#include <hip/hip_runtime.h>
#include <hip/hip_fp16.h>
#include <math.h>

typedef __attribute__((ext_vector_type(16))) _Float16 v16h;
typedef __attribute__((ext_vector_type(8)))  _Float16 v8h;
typedef __attribute__((ext_vector_type(8)))  float    v8f;

#define V_SIZE  8192
#define F_SIZE  256
#define B_SIZE  32
#define L_SIZE  64
#define NROWS   (B_SIZE * (L_SIZE - 1))   // 2016
#define MTILES  (NROWS / 16)              // 126
#define NTILES  (V_SIZE / 16)             // 512
#define KCHUNKS (F_SIZE / 32)             // 8
#define NWAVES  8

#define LOG2E 1.4426950408889634f
#define LN2   0.6931471805599453f

// Native base-2 transcendentals: v_exp_f32 / v_log_f32 are base-2 on AMDGPU.
#define EXP2F(x) __builtin_amdgcn_exp2f(x)
#define LOG2F(x) __builtin_amdgcn_logf(x)

// ---------------------------------------------------------------------------
// out[0] = B * log(V)  (nll_first term) — also initializes the accumulator.
// ---------------------------------------------------------------------------
__global__ void init_out_kernel(float* __restrict__ out) {
    out[0] = (float)B_SIZE * LN2 * LOG2F((float)V_SIZE);
}

// ---------------------------------------------------------------------------
// fwT[k][v] = (f16)(features[v][k] * w[1+k]);  s_feat[v] = sum_k fw.
// fwT stored transposed (F x V) so WMMA B-operand loads are contiguous.
// ---------------------------------------------------------------------------
__global__ void prep_kernel(const int*   __restrict__ features,
                            const float* __restrict__ weights,
                            _Float16*    __restrict__ fwT,
                            float*       __restrict__ s_feat) {
    const int v = blockIdx.x * blockDim.x + threadIdx.x;
    if (v >= V_SIZE) return;
    float s = 0.0f;
    for (int k = 0; k < F_SIZE; ++k) {
        const float fw = (float)features[v * F_SIZE + k] * weights[1 + k];
        s += fw;
        fwT[k * V_SIZE + v] = (_Float16)fw;
    }
    s_feat[v] = s;
}

// ---------------------------------------------------------------------------
// prevh[r][k] = (f16)features[seq[b, t]][k],  r = b*(L-1)+t.
// ---------------------------------------------------------------------------
__global__ void gather_kernel(const int* __restrict__ features,
                              const int* __restrict__ sequences,
                              _Float16*  __restrict__ prevh) {
    const int r = blockIdx.x;           // 0 .. NROWS-1
    const int k = threadIdx.x;          // 0 .. F-1
    const int b = r / (L_SIZE - 1);
    const int t = r % (L_SIZE - 1);
    const int idx = sequences[b * L_SIZE + t];
    prevh[r * F_SIZE + k] = (_Float16)features[idx * F_SIZE + k];
}

// ---------------------------------------------------------------------------
// One workgroup (8 waves) per 16-row M-tile. Each wave sweeps N-tiles strided
// by 8: K=256 via 8 chained v_wmma_f32_16x16x32_f16, then a BRANCH-FREE
// online logsumexp in the log2 domain (co-executes with the XDL pipe).
// ---------------------------------------------------------------------------
__global__ __launch_bounds__(256, 1) void gemm_nll_kernel(
        const _Float16* __restrict__ prevh,     // [NROWS][F]
        const _Float16* __restrict__ fwT,       // [F][V]
        const float*    __restrict__ s_feat,    // [V]
        const float*    __restrict__ weights,   // [F+1]
        const int*      __restrict__ sequences, // [B][L]
        float*          __restrict__ out) {
    __shared__ float red_mx[NWAVES * 32 * 8];
    __shared__ float red_sm[NWAVES * 32 * 8];
    __shared__ float row_nll[16];

    const int tid  = threadIdx.x;
    const int wave = tid >> 5;
    const int lane = tid & 31;
    const int mt   = blockIdx.x;
    const int m    = lane & 15;      // M-row (A) / N-col (C) owned by this lane
    const int half = lane >> 4;      // lane group (ISA 16-bit A layout)
    const float w0       = weights[0];
    const float scale    = w0 * LOG2E;        // logits tracked in log2 domain
    const float twoscale = 2.0f * scale;

    // --- Load A (16x256) into registers per the ISA 16-bit A 16x32 layout:
    //     lanes 0-15 hold K {0..7,16..23}, lanes 16-31 hold K {8..15,24..31}.
    v16h a[KCHUNKS];
    const _Float16* arow = prevh + (size_t)(mt * 16 + m) * F_SIZE;
    #pragma unroll
    for (int kb = 0; kb < KCHUNKS; ++kb) {
        const _Float16* base = arow + kb * 32;
        const v8h lo = *(const v8h*)(base + (half ? 8  : 0));
        const v8h hi = *(const v8h*)(base + (half ? 24 : 16));
        v16h av;
        #pragma unroll
        for (int i = 0; i < 8; ++i) { av[i] = lo[i]; av[8 + i] = hi[i]; }
        a[kb] = av;
    }

    float mx[8], sm[8];
    #pragma unroll
    for (int r = 0; r < 8; ++r) { mx[r] = -INFINITY; sm[r] = 0.0f; }

    for (int nt = wave; nt < NTILES; nt += NWAVES) {
        const int n0 = nt * 16;
        v8f c = {};
        #pragma unroll
        for (int kb = 0; kb < KCHUNKS; ++kb) {
            // B layout: lane = N col (n0+m); v16h = K {0..7,16..23} (half 0) / {8..15,24..31} (half 1).
            v16h bmat;
            #pragma unroll
            for (int i = 0; i < 16; ++i) {
                const int kk = kb * 32 + (i < 8 ? 8 * half + i : 16 + 8 * half + (i - 8));
                bmat[i] = fwT[(size_t)kk * V_SIZE + n0 + m];
            }
            c = __builtin_amdgcn_wmma_f32_16x16x32_f16(
                    /*neg_a=*/false, a[kb], /*neg_b=*/false, bmat,
                    /*c_mod=*/(short)0, c, /*reuse_a=*/false, /*reuse_b=*/false);
        }
        // Lane owns column n = n0+m for rows (half ? 8..15 : 0..7).
        const float sfs = s_feat[n0 + m] * scale;
        #pragma unroll
        for (int r = 0; r < 8; ++r) {
            const float y  = __builtin_fmaf(c[r], twoscale, -sfs); // log2-logit
            const float nm = fmaxf(mx[r], y);
            sm[r] = sm[r] * EXP2F(mx[r] - nm) + EXP2F(y - nm);     // branch-free
            mx[r] = nm;
        }
    }

    // --- Dump per-lane partials, merge per row.
    #pragma unroll
    for (int r = 0; r < 8; ++r) {
        red_mx[(wave * 32 + lane) * 8 + r] = mx[r];
        red_sm[(wave * 32 + lane) * 8 + r] = sm[r];
    }
    __syncthreads();

    if (tid < 16) {
        const int mrow  = tid;
        const int lbase = (mrow < 8) ? 0 : 16;
        const int rr    = mrow & 7;
        float M = -INFINITY, S = 0.0f;
        for (int w = 0; w < NWAVES; ++w) {
            for (int li = 0; li < 16; ++li) {
                const int idx = ((w * 32 + lbase + li) * 8 + rr);
                const float m2 = red_mx[idx];
                const float s2 = red_sm[idx];
                const float nm = fmaxf(M, m2);
                S = S * EXP2F(M - nm) + s2 * EXP2F(m2 - nm);
                M = nm;
            }
        }
        // Target logit for this row (per-row constants cancel vs logsumexp).
        const int R   = mt * 16 + mrow;
        const int b   = R / (L_SIZE - 1);
        const int t   = R % (L_SIZE - 1);
        const int tgt = sequences[b * L_SIZE + t + 1];
        const _Float16* ar = prevh + (size_t)R * F_SIZE;
        float dot = 0.0f;
        for (int k = 0; k < F_SIZE; ++k)
            dot += (float)ar[k] * (float)fwT[(size_t)k * V_SIZE + tgt];
        const float ytgt2 = __builtin_fmaf(dot, twoscale, -s_feat[tgt] * scale);
        row_nll[mrow] = LN2 * ((M + LOG2F(S)) - ytgt2);   // -log p(tgt)
    }
    __syncthreads();

    if (tid == 0) {
        float acc = 0.0f;
        #pragma unroll
        for (int i = 0; i < 16; ++i) acc += row_nll[i];
        atomicAdd(out, acc);
    }
}

// ---------------------------------------------------------------------------
extern "C" void kernel_launch(void* const* d_in, const int* in_sizes, int n_in,
                              void* d_out, int out_size, void* d_ws, size_t ws_size,
                              hipStream_t stream) {
    (void)in_sizes; (void)n_in; (void)out_size; (void)ws_size;

    const float* weights   = (const float*)d_in[0];   // (F+1,) f32
    const int*   features  = (const int*)d_in[1];     // (V, F) int32 {0,1}
    const int*   sequences = (const int*)d_in[2];     // (B, L) int32
    float*       out       = (float*)d_out;           // scalar

    char* ws = (char*)d_ws;
    _Float16* fwT    = (_Float16*)(ws);                                   // 4 MB
    _Float16* prevh  = (_Float16*)(ws + (size_t)F_SIZE * V_SIZE * 2);     // ~1 MB
    float*    s_feat = (float*)   (ws + (size_t)F_SIZE * V_SIZE * 2
                                      + (size_t)NROWS * F_SIZE * 2);      // 32 KB

    init_out_kernel<<<1, 1, 0, stream>>>(out);
    prep_kernel   <<<V_SIZE / 256, 256, 0, stream>>>(features, weights, fwT, s_feat);
    gather_kernel <<<NROWS, F_SIZE, 0, stream>>>(features, sequences, prevh);
    gemm_nll_kernel<<<MTILES, 256, 0, stream>>>(prevh, fwT, s_feat, weights,
                                                sequences, out);
}
